// SuperKimiDeltaAttention_4501125726457
// MI455X (gfx1250) — matrix operand order repaired and checked by reading, not compiled, not run on hardware
//
#include <hip/hip_runtime.h>
#include <cstddef>

// ---------------------------------------------------------------- types
typedef __bf16 bf16;
typedef __attribute__((ext_vector_type(16))) __bf16 v16bf;
typedef __attribute__((ext_vector_type(8)))  float  v8f;

// ---------------------------------------------------------------- sizes
#define B_    2
#define T_    4096
#define D_    2048
#define H_    16
#define K_    128
#define P_    2048
#define C_    64
#define NC_   64
#define NS_   16
#define CPS_  4
#define LAT_  128
#define BT_   8192          // B_*T_
#define BH_   32            // B_*H_
#define BHN_  2048          // B_*H_*NC_
#define KK_   16384         // K_*K_

// ---------------------------------------------------------------- WMMA helpers (wave32, gfx1250)
union FragU { v16bf v; unsigned u[8]; };

__device__ __forceinline__ v8f v8f_zero() {
  v8f z = {0.f,0.f,0.f,0.f,0.f,0.f,0.f,0.f};
  return z;
}

// A-operand: 16x32 bf16 tile, row-major (row = M), per ISA A-map:
// lanes 0-15 hold K 0..7 (V0-3) and 16..23 (V4-7); lanes 16-31 hold 8..15 / 24..31.
__device__ __forceinline__ v16bf frag_row(const bf16* base, int stride, int lane) {
  const bf16* row = base + (lane & 15) * stride;
  int hi = (lane >> 4) << 3;
  FragU f;
#pragma unroll
  for (int p = 0; p < 8; ++p) {
    int k = 2*p + ((p >= 4) ? 8 : 0) + hi;
    f.u[p] = *(const unsigned*)(row + k);
  }
  return f.v;
}

// B-operand from W stored row-major N x Kd (computing A @ W^T), per ISA B-map:
// lanes 0-15 hold K 0..7 and 16..23; lanes 16-31 hold 8..15 / 24..31.
__device__ __forceinline__ v16bf frag_col(const bf16* base, int stride, int lane) {
  const bf16* row = base + (lane & 15) * stride;
  int hi = (lane >> 4) << 3;
  FragU f;
#pragma unroll
  for (int p = 0; p < 8; ++p) {
    int k = 2*p + ((p >= 4) ? 8 : 0) + hi;
    f.u[p] = *(const unsigned*)(row + k);
  }
  return f.v;
}

__device__ __forceinline__ v8f wmma_bf16(v16bf a, v16bf b, v8f c) {
  return __builtin_amdgcn_wmma_f32_16x16x32_bf16(false, a, false, b, (short)0, c, false, false);
}

// ---------------------------------------------------------------- cast
__global__ void k_cast_bf16(const float* __restrict__ src, bf16* __restrict__ dst, long n) {
  long i = (long)blockIdx.x * blockDim.x + threadIdx.x;
  if (i < n) dst[i] = (bf16)src[i];
}

// ---------------------------------------------------------------- generic GEMM: C[M,N] = A[M,Kd] @ W[N,Kd]^T
// 256 threads = 8 waves; block tile 128x128, K-step 64. Kd must be a multiple of 64.
#define SBK 72  // padded LDS row stride (bf16), 144B => 16B aligned rows, even for pair reads
__global__ __launch_bounds__(256) void k_gemm_bf16(const bf16* __restrict__ A,
                                                   const bf16* __restrict__ W,
                                                   float* __restrict__ Cmat,
                                                   int M, int N, int Kd) {
  __shared__ bf16 sA[128 * SBK];
  __shared__ bf16 sB[128 * SBK];
  int tid = threadIdx.x, lane = tid & 31, wave = tid >> 5;
  int wr = wave >> 2, wc = wave & 3;           // 2x4 waves, each 4x2 tiles of 16x16
  int bm = blockIdx.y * 128, bn = blockIdx.x * 128;

  v8f acc[4][2];
#pragma unroll
  for (int i = 0; i < 4; ++i) { acc[i][0] = v8f_zero(); acc[i][1] = v8f_zero(); }

  for (int k0 = 0; k0 < Kd; k0 += 64) {
#pragma unroll
    for (int i = 0; i < 4; ++i) {              // 1024 uint4 per tile, 4 per thread
      int idx = tid + i * 256;
      int row = idx >> 3, c8 = idx & 7;
      uint4 va = make_uint4(0u, 0u, 0u, 0u);
      int gm = bm + row;
      if (gm < M) va = *(const uint4*)(A + (size_t)gm * Kd + k0 + c8 * 8);
      *(uint4*)(sA + row * SBK + c8 * 8) = va;
      uint4 vb = make_uint4(0u, 0u, 0u, 0u);
      int gn = bn + row;
      if (gn < N) vb = *(const uint4*)(W + (size_t)gn * Kd + k0 + c8 * 8);
      *(uint4*)(sB + row * SBK + c8 * 8) = vb;
    }
    __syncthreads();

    if (k0 + 64 < Kd) {                        // emits global_prefetch_b8
      __builtin_prefetch(A + (size_t)(bm + (tid >> 1)) * Kd + k0 + 64, 0, 3);
      __builtin_prefetch(W + (size_t)(bn + (tid >> 1)) * Kd + k0 + 64, 0, 3);
    }

#pragma unroll
    for (int kk = 0; kk < 64; kk += 32) {
      v16bf b0 = frag_col(sB + (wc * 2 + 0) * 16 * SBK + kk, SBK, lane);
      v16bf b1 = frag_col(sB + (wc * 2 + 1) * 16 * SBK + kk, SBK, lane);
#pragma unroll
      for (int tr = 0; tr < 4; ++tr) {
        v16bf a = frag_row(sA + (wr * 4 + tr) * 16 * SBK + kk, SBK, lane);
        acc[tr][0] = wmma_bf16(a, b0, acc[tr][0]);
        acc[tr][1] = wmma_bf16(a, b1, acc[tr][1]);
      }
    }
    __syncthreads();
  }

#pragma unroll
  for (int tr = 0; tr < 4; ++tr)
#pragma unroll
    for (int tc = 0; tc < 2; ++tc) {
      int m0 = bm + (wr * 4 + tr) * 16;
      int n0 = bn + (wc * 2 + tc) * 16;
      int nn = n0 + (lane & 15);
      if (nn < N) {
#pragma unroll
        for (int r = 0; r < 8; ++r) {
          int mm = m0 + r + ((lane >> 4) << 3);
          if (mm < M) Cmat[(size_t)mm * N + nn] = acc[tr][tc][r];
        }
      }
    }
}

// ---------------------------------------------------------------- causal depthwise conv4 + SiLU
__global__ void k_dwconv_silu(const float* __restrict__ pre, const float* __restrict__ w,
                              float* __restrict__ post) {
  long i = (long)blockIdx.x * blockDim.x + threadIdx.x;
  if (i >= (long)BT_ * P_) return;
  int p = (int)(i % P_);
  long bt = i / P_;
  int t = (int)(bt % T_);
  long brow = bt - t;  // b*T
  float s = 0.f;
#pragma unroll
  for (int j = 0; j < 4; ++j) {
    int tt = t - 3 + j;
    if (tt >= 0) s += w[p * 4 + j] * pre[(brow + tt) * P_ + p];
  }
  post[i] = s / (1.f + expf(-s));  // silu
}

// ---------------------------------------------------------------- per-(b,t,h) L2 norm over K, in place
__global__ void k_l2norm(float* __restrict__ x, float scale) {
  __shared__ float red[128];
  long row = blockIdx.x;
  int lane = threadIdx.x;
  float v = x[row * 128 + lane];
  red[lane] = v * v;
  __syncthreads();
  for (int s = 64; s > 0; s >>= 1) { if (lane < s) red[lane] += red[lane + s]; __syncthreads(); }
  float inv = rsqrtf(red[0] + 1e-12f) * scale;
  x[row * 128 + lane] = v * inv;
}

// ---------------------------------------------------------------- chunk prep: decay scan + operand packing
__global__ void k_prep(const float* __restrict__ q, const float* __restrict__ kx,
                       const float* __restrict__ v, const float* __restrict__ g_raw,
                       const float* __restrict__ b_logit, const float* __restrict__ A_log,
                       const float* __restrict__ dt_bias,
                       bf16* __restrict__ qt_bf, bf16* __restrict__ um_bf,
                       bf16* __restrict__ umT_bf, bf16* __restrict__ bw_bf,
                       float* __restrict__ bw_f, float* __restrict__ bv_f,
                       float* __restrict__ wl) {
  int i = blockIdx.x * blockDim.x + threadIdx.x;  // (b,h,n,k)
  if (i >= BH_ * NC_ * K_) return;
  int kk = i & 127;
  int r = i >> 7;
  int n = r & 63; r >>= 6;
  int h = r & 15;
  int b = r >> 4;

  float a  = -expf(A_log[h]);
  float db = dt_bias[h * K_ + kk];
  float cum = 0.f;
  long chunkBase = ((long)((b * H_ + h) * NC_ + n)) * (C_ * K_);
  long wlBase    = ((long)((b * H_ + h) * NC_ + n)) * K_;

  for (int c = 0; c < C_; ++c) {
    long t  = (long)n * C_ + c;
    long xi = ((long)(b * T_ + t) * H_ + h) * K_ + kk;
    float gr = g_raw[xi] + db;
    float sp = (gr > 20.f) ? gr : log1pf(expf(gr));   // softplus
    cum += a * sp;
    float cc = fmaxf(cum, -15.f);
    float wg = expf(cc), wi = expf(-cc);
    float qv = q[xi], kv = kx[xi], vv = v[xi];
    float bet = 1.f / (1.f + expf(-b_logit[(b * T_ + t) * H_ + h]));
    float qt = qv * wg, um = kv * wi, bw = bet * kv * wg, bv = bet * vv;
    long o = chunkBase + (long)c * K_ + kk;
    qt_bf[o] = (bf16)qt;
    um_bf[o] = (bf16)um;
    bw_bf[o] = (bf16)bw;
    bw_f[o]  = bw;
    bv_f[o]  = bv;
    umT_bf[chunkBase + (long)kk * C_ + c] = (bf16)um;
    if (c == C_ - 1) wl[wlBase + kk] = wg;
  }
}

// ---------------------------------------------------------------- per-chunk: L, intra (WMMA) + unit-lower solves
// dynamic LDS: bw(16K) um(16K) qt(16K) L(16K f32) ev(32K f32) ew(32K f32) = 128K
__global__ __launch_bounds__(256) void k_lsolve(const bf16* __restrict__ bw_bf,
                                                const bf16* __restrict__ um_bf,
                                                const bf16* __restrict__ qt_bf,
                                                const float* __restrict__ bw_f,
                                                const float* __restrict__ bv_f,
                                                float* __restrict__ ev_f,
                                                bf16* __restrict__ ew_bf,
                                                bf16* __restrict__ intra_bf) {
  extern __shared__ char smem[];
  bf16*  sBW = (bf16*)smem;                 // 64*128
  bf16*  sUM = sBW + C_ * K_;
  bf16*  sQT = sUM + C_ * K_;
  float* sL  = (float*)(sQT + C_ * K_);     // 64*64
  float* sEV = sL + C_ * C_;                // 64*128
  float* sEW = sEV + C_ * K_;               // 64*128

  int tid = threadIdx.x, lane = tid & 31, wave = tid >> 5;
  long cb = (long)blockIdx.x * (C_ * K_);
  long ib = (long)blockIdx.x * (C_ * C_);

  for (int i = tid; i < (C_ * K_) / 8; i += 256) {   // 1024 uint4 each
    ((uint4*)sBW)[i] = ((const uint4*)(bw_bf + cb))[i];
    ((uint4*)sUM)[i] = ((const uint4*)(um_bf + cb))[i];
    ((uint4*)sQT)[i] = ((const uint4*)(qt_bf + cb))[i];
  }
  for (int i = tid; i < C_ * K_; i += 256) {
    sEV[i] = bv_f[cb + i];
    sEW[i] = bw_f[cb + i];
  }
  __syncthreads();

  // L = tril(bw @ um^T, -1), intra = tril(qt @ um^T, 0)
  for (int t = wave; t < 16; t += 8) {
    int tm = t >> 2, tn = t & 3;
    v8f accL = v8f_zero(), accI = v8f_zero();
#pragma unroll
    for (int k0 = 0; k0 < K_; k0 += 32) {
      v16bf bb = frag_col(sUM + tn * 16 * K_ + k0, K_, lane);
      accL = wmma_bf16(frag_row(sBW + tm * 16 * K_ + k0, K_, lane), bb, accL);
      accI = wmma_bf16(frag_row(sQT + tm * 16 * K_ + k0, K_, lane), bb, accI);
    }
    int nn = tn * 16 + (lane & 15);
#pragma unroll
    for (int r = 0; r < 8; ++r) {
      int mm = tm * 16 + r + ((lane >> 4) << 3);
      sL[mm * C_ + nn] = (mm > nn) ? accL[r] : 0.f;
      intra_bf[ib + mm * C_ + nn] = (bf16)((mm >= nn) ? accI[r] : 0.f);
    }
  }
  __syncthreads();

  // forward substitution, unit diagonal: two RHS blocks (ev | ew), 128 columns each
  float* buf = (tid < 128) ? sEV : sEW;
  int vcol = tid & 127;
  for (int i = 1; i < C_; ++i) {
    float s = 0.f;
    for (int j = 0; j < i; ++j) s += sL[i * C_ + j] * buf[j * K_ + vcol];
    buf[i * K_ + vcol] -= s;
    __syncthreads();
  }
  for (int i = tid; i < C_ * K_; i += 256) {
    ev_f[cb + i] = sEV[i];
    ew_bf[cb + i] = (bf16)sEW[i];
  }
}

// ---------------------------------------------------------------- sequential chunk scan, state in registers
// dynamic LDS: St_bf(32K) qt(16K) ew(16K) umT(16K) intra(8K) wl(512) deltaT(16K) = 104.5K
__global__ __launch_bounds__(256) void k_scan(const bf16* __restrict__ qt_bf,
                                              const bf16* __restrict__ ew_bf,
                                              const bf16* __restrict__ umT_bf,
                                              const bf16* __restrict__ intra_bf,
                                              const float* __restrict__ ev_f,
                                              const float* __restrict__ wl,
                                              float* __restrict__ out_f,
                                              float* __restrict__ s_snap) {
  extern __shared__ char smem[];
  bf16*  sSt  = (bf16*)smem;                 // [v=128][k=128]  == S^T
  bf16*  sQT  = sSt + 128 * 128;             // [c=64][k=128]
  bf16*  sEW  = sQT + C_ * K_;
  bf16*  sUMT = sEW + C_ * K_;               // [k=128][c=64]
  bf16*  sIN  = sUMT + K_ * C_;              // [c=64][j=64]
  float* sWL  = (float*)(sIN + C_ * C_);     // [k=128]
  bf16*  sDT  = (bf16*)(sWL + K_);           // [v=128][c=64]  == delta^T

  int tid = threadIdx.x, lane = tid & 31, wave = tid >> 5;
  int h = blockIdx.x & 15, b = blockIdx.x >> 4;

  v8f st[8];                                  // wave owns v-rows [16w,16w+16), all 8 k-tiles
#pragma unroll
  for (int t = 0; t < 8; ++t) st[t] = v8f_zero();
  for (int i = tid; i < 128 * 128; i += 256) sSt[i] = (bf16)0.0f;
  __syncthreads();

  for (int n = 0; n < NC_; ++n) {
    long cb = ((long)((b * H_ + h) * NC_ + n)) * (C_ * K_);
    long ib = ((long)((b * H_ + h) * NC_ + n)) * (C_ * C_);
    long wb = ((long)((b * H_ + h) * NC_ + n)) * K_;

    for (int i = tid; i < (C_ * K_) / 8; i += 256) {
      ((uint4*)sQT)[i]  = ((const uint4*)(qt_bf + cb))[i];
      ((uint4*)sEW)[i]  = ((const uint4*)(ew_bf + cb))[i];
      ((uint4*)sUMT)[i] = ((const uint4*)(umT_bf + cb))[i];
    }
    for (int i = tid; i < (C_ * C_) / 8; i += 256)
      ((uint4*)sIN)[i] = ((const uint4*)(intra_bf + ib))[i];
    for (int i = tid; i < K_; i += 256) sWL[i] = wl[wb + i];
    __syncthreads();

    // delta = ev - ew @ S   (write delta^T to LDS as bf16)
    for (int t = wave; t < 32; t += 8) {
      int tm = t >> 3, tn = t & 7;
      v8f acc = v8f_zero();
#pragma unroll
      for (int k0 = 0; k0 < K_; k0 += 32)
        acc = wmma_bf16(frag_row(sEW + tm * 16 * K_ + k0, K_, lane),
                        frag_col(sSt + tn * 16 * K_ + k0, K_, lane), acc);
      int nn = tn * 16 + (lane & 15);
#pragma unroll
      for (int r = 0; r < 8; ++r) {
        int mm = tm * 16 + r + ((lane >> 4) << 3);
        float d = ev_f[cb + mm * K_ + nn] - acc[r];
        sDT[nn * C_ + mm] = (bf16)d;
      }
    }
    __syncthreads();

    // o = qt @ S + intra @ delta  (uses old S)
    for (int t = wave; t < 32; t += 8) {
      int tm = t >> 3, tn = t & 7;
      v8f acc = v8f_zero();
#pragma unroll
      for (int k0 = 0; k0 < K_; k0 += 32)
        acc = wmma_bf16(frag_row(sQT + tm * 16 * K_ + k0, K_, lane),
                        frag_col(sSt + tn * 16 * K_ + k0, K_, lane), acc);
#pragma unroll
      for (int k0 = 0; k0 < C_; k0 += 32)
        acc = wmma_bf16(frag_row(sIN + tm * 16 * C_ + k0, C_, lane),
                        frag_col(sDT + tn * 16 * C_ + k0, C_, lane), acc);
      int nn = tn * 16 + (lane & 15);
#pragma unroll
      for (int r = 0; r < 8; ++r) {
        int mm = tm * 16 + r + ((lane >> 4) << 3);
        long tg = (long)n * C_ + mm;
        out_f[((long)(b * T_ + tg) * H_ + h) * K_ + nn] = acc[r];
      }
    }

    // S^T update: St[v][k] = wl[k] * (St[v][k] + sum_c deltaT[v][c]*umT[k][c])
    for (int tn = 0; tn < 8; ++tn) {
      v8f acc = v8f_zero();
#pragma unroll
      for (int k0 = 0; k0 < C_; k0 += 32)
        acc = wmma_bf16(frag_row(sDT + wave * 16 * C_ + k0, C_, lane),
                        frag_col(sUMT + tn * 16 * C_ + k0, C_, lane), acc);
      int kcol = tn * 16 + (lane & 15);
      float wlv = sWL[kcol];
#pragma unroll
      for (int r = 0; r < 8; ++r) st[tn][r] = wlv * (st[tn][r] + acc[r]);
    }
    __syncthreads();                          // everyone done reading old sSt

#pragma unroll
    for (int tn = 0; tn < 8; ++tn) {
      int kcol = tn * 16 + (lane & 15);
#pragma unroll
      for (int r = 0; r < 8; ++r) {
        int vrow = wave * 16 + r + ((lane >> 4) << 3);
        sSt[vrow * K_ + kcol] = (bf16)st[tn][r];
      }
    }
    __syncthreads();

    if ((n & (CPS_ - 1)) == CPS_ - 1) {       // snapshot S (k-major)
      int s = n >> 2;
      long sb = ((long)(s * B_ + b) * H_ + h) * KK_;
#pragma unroll
      for (int tn = 0; tn < 8; ++tn) {
        int kcol = tn * 16 + (lane & 15);
#pragma unroll
        for (int r = 0; r < 8; ++r) {
          int vrow = wave * 16 + r + ((lane >> 4) << 3);
          s_snap[sb + (long)kcol * K_ + vrow] = st[tn][r];
        }
      }
    }
  }
}

// ---------------------------------------------------------------- rmsnorm(out)*sigmoid(gate) -> bf16
__global__ void k_outgate(const float* __restrict__ out_f, const float* __restrict__ gate_raw,
                          const float* __restrict__ bgb, const float* __restrict__ onw,
                          bf16* __restrict__ obf) {
  __shared__ float red[128];
  long row = blockIdx.x;                      // (b*T+t)*H + h
  int lane = threadIdx.x;
  long base = row * 128 + lane;
  float v = out_f[base];
  red[lane] = v * v;
  __syncthreads();
  for (int s = 64; s > 0; s >>= 1) { if (lane < s) red[lane] += red[lane + s]; __syncthreads(); }
  float r = rsqrtf(red[0] / 128.f + 1e-6f);
  int h = (int)(row % H_);
  float g = gate_raw[base] + bgb[h * 128 + lane];
  float sg = 1.f / (1.f + expf(-g));
  obf[base] = (bf16)(v * r * onw[lane] * sg);
}

// ---------------------------------------------------------------- snapshot rmsnorm rows (len 16384) -> bf16
__global__ __launch_bounds__(256) void k_snaprms(const float* __restrict__ s_snap,
                                                 const float* __restrict__ w,
                                                 bf16* __restrict__ snap_bf) {
  __shared__ float red[256];
  int row = blockIdx.x;                       // (s*B+b)*H + h, 512 rows
  int tid = threadIdx.x;
  const float* src = s_snap + (long)row * KK_;
  float acc = 0.f;
  for (int i = tid; i < KK_; i += 256) { float v = src[i]; acc += v * v; }
  red[tid] = acc;
  __syncthreads();
  for (int s = 128; s > 0; s >>= 1) { if (tid < s) red[tid] += red[tid + s]; __syncthreads(); }
  float r = rsqrtf(red[0] / (float)KK_ + 1e-6f);
  bf16* dst = snap_bf + (long)row * KK_;
  for (int i = tid; i < KK_; i += 256) dst[i] = (bf16)(src[i] * r * w[i]);
}

// ---------------------------------------------------------------- (s,b,h,l) -> (b,s,h,l)
__global__ void k_snap_permute(const float* __restrict__ snap_y, float* __restrict__ out2) {
  int i = blockIdx.x * blockDim.x + threadIdx.x;  // NS*B*H*LAT = 65536
  if (i >= NS_ * B_ * H_ * LAT_) return;
  int l = i & 127;
  int r = i >> 7;
  int h = r & 15; r >>= 4;
  int b = r & 1;
  int s = r >> 1;
  out2[((long)(b * NS_ + s) * H_ + h) * LAT_ + l] = snap_y[i];
}

// ---------------------------------------------------------------- host helpers
static void gemm_launch(const bf16* A, const bf16* W, float* Cm, int M, int N, int Kd,
                        hipStream_t s) {
  dim3 grid((N + 127) / 128, (M + 127) / 128);
  k_gemm_bf16<<<grid, 256, 0, s>>>(A, W, Cm, M, N, Kd);
}
static void cast_launch(const float* src, bf16* dst, long n, hipStream_t s) {
  k_cast_bf16<<<(unsigned)((n + 255) / 256), 256, 0, s>>>(src, dst, n);
}

// ---------------------------------------------------------------- entry
extern "C" void kernel_launch(void* const* d_in, const int* in_sizes, int n_in,
                              void* d_out, int out_size, void* d_ws, size_t ws_size,
                              hipStream_t stream) {
  (void)in_sizes; (void)n_in; (void)out_size; (void)ws_size;
  const float* x        = (const float*)d_in[0];
  const float* Wq       = (const float*)d_in[1];
  const float* Wk       = (const float*)d_in[2];
  const float* Wv       = (const float*)d_in[3];
  const float* conv_q   = (const float*)d_in[4];
  const float* conv_k   = (const float*)d_in[5];
  const float* conv_v   = (const float*)d_in[6];
  const float* A_log    = (const float*)d_in[7];
  const float* dt_bias  = (const float*)d_in[8];
  const float* Wfa      = (const float*)d_in[9];
  const float* Wfb      = (const float*)d_in[10];
  const float* Wb       = (const float*)d_in[11];
  const float* Wga      = (const float*)d_in[12];
  const float* Wgb      = (const float*)d_in[13];
  const float* bgb      = (const float*)d_in[14];
  const float* o_norm_w = (const float*)d_in[15];
  const float* Wo       = (const float*)d_in[16];
  const float* snap_nw  = (const float*)d_in[17];
  const float* snap_dn  = (const float*)d_in[18];

  char* wp = (char*)d_ws;
  auto alloc = [&](size_t bytes) -> void* {
    void* p = (void*)wp;
    wp += (bytes + 255) & ~(size_t)255;
    return p;
  };

  // bf16 copies
  bf16* xb   = (bf16*)alloc((size_t)BT_ * D_ * 2);
  bf16* WqB  = (bf16*)alloc((size_t)P_ * D_ * 2);
  bf16* WkB  = (bf16*)alloc((size_t)P_ * D_ * 2);
  bf16* WvB  = (bf16*)alloc((size_t)P_ * D_ * 2);
  bf16* WfaB = (bf16*)alloc((size_t)K_ * D_ * 2);
  bf16* WfbB = (bf16*)alloc((size_t)P_ * K_ * 2);
  bf16* WbB  = (bf16*)alloc((size_t)H_ * D_ * 2);
  bf16* WgaB = (bf16*)alloc((size_t)K_ * D_ * 2);
  bf16* WgbB = (bf16*)alloc((size_t)P_ * K_ * 2);
  bf16* WoB  = (bf16*)alloc((size_t)D_ * P_ * 2);
  bf16* SdB  = (bf16*)alloc((size_t)LAT_ * KK_ * 2);
  // fp32 activations
  float* q_pre  = (float*)alloc((size_t)BT_ * P_ * 4);
  float* k_pre  = (float*)alloc((size_t)BT_ * P_ * 4);
  float* v_pre  = (float*)alloc((size_t)BT_ * P_ * 4);
  float* q_post = (float*)alloc((size_t)BT_ * P_ * 4);
  float* k_post = (float*)alloc((size_t)BT_ * P_ * 4);
  float* v_post = (float*)alloc((size_t)BT_ * P_ * 4);
  float* f_lat  = (float*)alloc((size_t)BT_ * K_ * 4);
  float* g_lat  = (float*)alloc((size_t)BT_ * K_ * 4);
  bf16*  f_latB = (bf16*)alloc((size_t)BT_ * K_ * 2);
  bf16*  g_latB = (bf16*)alloc((size_t)BT_ * K_ * 2);
  float* g_raw  = (float*)alloc((size_t)BT_ * P_ * 4);
  float* gateR  = (float*)alloc((size_t)BT_ * P_ * 4);
  float* b_log  = (float*)alloc((size_t)BT_ * H_ * 4);
  // chunk operands
  bf16*  qtB  = (bf16*)alloc((size_t)BHN_ * C_ * K_ * 2);
  bf16*  umB  = (bf16*)alloc((size_t)BHN_ * C_ * K_ * 2);
  bf16*  umTB = (bf16*)alloc((size_t)BHN_ * C_ * K_ * 2);
  bf16*  bwB  = (bf16*)alloc((size_t)BHN_ * C_ * K_ * 2);
  float* bwF  = (float*)alloc((size_t)BHN_ * C_ * K_ * 4);
  float* bvF  = (float*)alloc((size_t)BHN_ * C_ * K_ * 4);
  float* evF  = (float*)alloc((size_t)BHN_ * C_ * K_ * 4);
  bf16*  ewB  = (bf16*)alloc((size_t)BHN_ * C_ * K_ * 2);
  bf16*  inB  = (bf16*)alloc((size_t)BHN_ * C_ * C_ * 2);
  float* wlF  = (float*)alloc((size_t)BHN_ * K_ * 4);
  // scan outputs
  float* outF  = (float*)alloc((size_t)BT_ * P_ * 4);
  float* sSnap = (float*)alloc((size_t)NS_ * B_ * H_ * KK_ * 4);
  bf16*  snapB = (bf16*)alloc((size_t)NS_ * B_ * H_ * KK_ * 2);
  float* snapY = (float*)alloc((size_t)NS_ * B_ * H_ * LAT_ * 4);
  bf16*  oB    = (bf16*)alloc((size_t)BT_ * P_ * 2);

  // dynamic-LDS opt-in (>64KB)
  (void)hipFuncSetAttribute((const void*)k_lsolve,
                            hipFuncAttributeMaxDynamicSharedMemorySize, 131072);
  (void)hipFuncSetAttribute((const void*)k_scan,
                            hipFuncAttributeMaxDynamicSharedMemorySize, 107008);

  // 1) bf16 casts
  cast_launch(x,       xb,   (long)BT_ * D_, stream);
  cast_launch(Wq,      WqB,  (long)P_ * D_,  stream);
  cast_launch(Wk,      WkB,  (long)P_ * D_,  stream);
  cast_launch(Wv,      WvB,  (long)P_ * D_,  stream);
  cast_launch(Wfa,     WfaB, (long)K_ * D_,  stream);
  cast_launch(Wfb,     WfbB, (long)P_ * K_,  stream);
  cast_launch(Wb,      WbB,  (long)H_ * D_,  stream);
  cast_launch(Wga,     WgaB, (long)K_ * D_,  stream);
  cast_launch(Wgb,     WgbB, (long)P_ * K_,  stream);
  cast_launch(Wo,      WoB,  (long)D_ * P_,  stream);
  cast_launch(snap_dn, SdB,  (long)LAT_ * KK_, stream);

  // 2) projections
  gemm_launch(xb, WqB,  q_pre, BT_, P_, D_, stream);
  gemm_launch(xb, WkB,  k_pre, BT_, P_, D_, stream);
  gemm_launch(xb, WvB,  v_pre, BT_, P_, D_, stream);
  gemm_launch(xb, WbB,  b_log, BT_, H_, D_, stream);
  gemm_launch(xb, WfaB, f_lat, BT_, K_, D_, stream);
  gemm_launch(xb, WgaB, g_lat, BT_, K_, D_, stream);

  // 3) depthwise conv + silu
  unsigned convGrid = (unsigned)(((long)BT_ * P_ + 255) / 256);
  k_dwconv_silu<<<convGrid, 256, 0, stream>>>(q_pre, conv_q, q_post);
  k_dwconv_silu<<<convGrid, 256, 0, stream>>>(k_pre, conv_k, k_post);
  k_dwconv_silu<<<convGrid, 256, 0, stream>>>(v_pre, conv_v, v_post);

  // 4) low-rank second stages
  cast_launch(f_lat, f_latB, (long)BT_ * K_, stream);
  cast_launch(g_lat, g_latB, (long)BT_ * K_, stream);
  gemm_launch(f_latB, WfbB, g_raw, BT_, P_, K_, stream);
  gemm_launch(g_latB, WgbB, gateR, BT_, P_, K_, stream);

  // 5) q/k L2 norm (q also scaled by K^-1/2)
  k_l2norm<<<BT_ * H_, 128, 0, stream>>>(q_post, 0.08838834764831845f);
  k_l2norm<<<BT_ * H_, 128, 0, stream>>>(k_post, 1.0f);

  // 6) decay scan + operand packing
  k_prep<<<(BH_ * NC_ * K_) / 256, 256, 0, stream>>>(q_post, k_post, v_post, g_raw, b_log,
                                                     A_log, dt_bias, qtB, umB, umTB, bwB,
                                                     bwF, bvF, wlF);

  // 7) per-chunk L/intra + triangular solves
  k_lsolve<<<BHN_, 256, 131072, stream>>>(bwB, umB, qtB, bwF, bvF, evF, ewB, inB);

  // 8) sequential scan per (b,h)
  k_scan<<<BH_, 256, 107008, stream>>>(qtB, ewB, umTB, inB, evF, wlF, outF, sSnap);

  // 9) gated rmsnorm -> bf16, final projection into d_out
  k_outgate<<<BT_ * H_, 128, 0, stream>>>(outF, gateR, bgb, o_norm_w, oB);
  gemm_launch(oB, WoB, (float*)d_out, BT_, D_, P_, stream);

  // 10) snapshots: rmsnorm -> bf16 -> down-projection -> permute into d_out tail
  k_snaprms<<<NS_ * B_ * H_, 256, 0, stream>>>(sSnap, snap_nw, snapB);
  gemm_launch(snapB, SdB, snapY, NS_ * B_ * H_, LAT_, KK_, stream);
  float* out2 = (float*)d_out + (size_t)BT_ * D_;
  k_snap_permute<<<(NS_ * B_ * H_ * LAT_) / 256, 256, 0, stream>>>(snapY, out2);
}
